// CP_B_53669911331094
// MI455X (gfx1250) — compile-verified, matrix operand order checked by reading, not run on hardware
//
#include <hip/hip_runtime.h>
#include <hip/hip_bf16.h>
#include <cstdint>

// ---------------------------------------------------------------------------
// CDNA5 / gfx1250 fused pipeline for the tensor-factorization MLP.
// Layer-1 GEMM (51.5 GFLOP): v_wmma_f32_16x16x32_bf16, A built on the fly
// with v_pk_mul_bf16 from LDS projections, B (W1 bf16) staged into LDS with
// double-buffered global_load_async_to_lds_b128 (ASYNCcnt) shared by 4 waves.
// ---------------------------------------------------------------------------

typedef __attribute__((ext_vector_type(16))) __bf16 v16bf;
typedef __attribute__((ext_vector_type(8)))  float  v8f;

#define BATCH  8192
#define LATENT 128
#define RED    64
#define IN_DIM 12288   // 3 * 64 * 64
#define H1_DIM 256
#define H2_DIM 128
#define PITCH  72      // bf16 elements per LDS row (16B-aligned, conflict-free)
#define NIT    (IN_DIM / 64)   // 192 K-iterations of 64

__device__ __forceinline__ unsigned int f32_to_bf16(float f) {
  unsigned int u = __float_as_uint(f);
  u += 0x7FFFu + ((u >> 16) & 1u);      // round-to-nearest-even
  return u >> 16;
}

// packed bf16 multiply: d.lo = a.lo*b.lo, d.hi = a.hi*b.hi  (VOP3P)
__device__ __forceinline__ unsigned int pk_mul_bf16(unsigned int a, unsigned int b) {
  unsigned int d;
  asm("v_pk_mul_bf16 %0, %1, %2" : "=v"(d) : "v"(a), "v"(b));
  return d;
}

union BFrag {
  v16bf        v;
  unsigned int u[8];
  uint4        q[2];
};

// ---------------------------------------------------------------------------
// Kernel 1: W1 fp32 -> bf16 (one-shot; 6.3 MB, L2-resident afterwards)
// ---------------------------------------------------------------------------
__global__ __launch_bounds__(256) void k_w1_to_bf16(
    const float* __restrict__ w1, unsigned short* __restrict__ w1bf, int n) {
  int i = blockIdx.x * 256 + threadIdx.x;
  if (i < n) w1bf[i] = (unsigned short)f32_to_bf16(w1[i]);
}

// ---------------------------------------------------------------------------
// Kernel 2: fused gather + 3 projections (L=128 -> R=64) + bias -> bf16.
// ---------------------------------------------------------------------------
__global__ __launch_bounds__(64) void k_gather_proj(
    const int* __restrict__ i_idx, const int* __restrict__ j_idx,
    const int* __restrict__ k_idx,
    const float* __restrict__ uemb, const float* __restrict__ iemb,
    const float* __restrict__ temb,
    const float* __restrict__ Wi, const float* __restrict__ bi,
    const float* __restrict__ Wj, const float* __restrict__ bj,
    const float* __restrict__ Wk, const float* __restrict__ bk,
    unsigned short* __restrict__ pi, unsigned short* __restrict__ pj,
    unsigned short* __restrict__ pk) {
  __shared__ float e[3][LATENT];
  const int b = blockIdx.x;
  const int t = threadIdx.x;
  const size_t ri = (size_t)i_idx[b] * LATENT;
  const size_t rj = (size_t)j_idx[b] * LATENT;
  const size_t rk = (size_t)k_idx[b] * LATENT;
  e[0][t] = uemb[ri + t];  e[0][t + 64] = uemb[ri + t + 64];
  e[1][t] = iemb[rj + t];  e[1][t + 64] = iemb[rj + t + 64];
  e[2][t] = temb[rk + t];  e[2][t + 64] = temb[rk + t + 64];
  __syncthreads();
  float a0 = bi[t], a1 = bj[t], a2 = bk[t];
  const float* wi = Wi + t * LATENT;
  const float* wj = Wj + t * LATENT;
  const float* wk = Wk + t * LATENT;
#pragma unroll 8
  for (int k = 0; k < LATENT; ++k) {
    a0 = fmaf(e[0][k], wi[k], a0);
    a1 = fmaf(e[1][k], wj[k], a1);
    a2 = fmaf(e[2][k], wk[k], a2);
  }
  pi[(size_t)b * RED + t] = (unsigned short)f32_to_bf16(a0);
  pj[(size_t)b * RED + t] = (unsigned short)f32_to_bf16(a1);
  pk[(size_t)b * RED + t] = (unsigned short)f32_to_bf16(a2);
}

// ---------------------------------------------------------------------------
// Kernel 3: h1 = relu(feats @ W1^T + b1) via bf16 WMMA.
//   Block: 128 threads (4 waves) = 64 batch rows x 128 cols; waves 2x2, each
//   owning 32x64 of C (2x4 tiles of 16x16). Grid 256.
//   W1 slab (128 cols x 64 K bf16 = 16KB) double-buffered in LDS via
//   global_load_async_to_lds_b128; all 4 waves consume it (4x traffic cut).
//   Per iter: wait_asynccnt -> barrier -> issue next slab -> compute.
// ---------------------------------------------------------------------------
__global__ __launch_bounds__(128) void k_layer1_wmma(
    const unsigned short* __restrict__ proj, // [3][BATCH][RED] bf16
    const unsigned short* __restrict__ w1bf, // [H1_DIM][IN_DIM] bf16
    const float* __restrict__ b1,            // [H1_DIM]
    float* __restrict__ h1)                  // [BATCH][H1_DIM]
{
  __shared__ unsigned short sp[3][64][PITCH];     // projections: 27.6 KB
  __shared__ unsigned short sb[2][128][PITCH];    // W1 slabs:    36.9 KB

  const int tid      = threadIdx.x;
  const int mb       = blockIdx.x >> 1;
  const int nb       = blockIdx.x & 1;
  const int row_base = mb * 64;
  const int col_base = nb * 128;

  // ---- load projections for 64 rows as dwords: 3*64*32 dwords ----
  {
    const unsigned int* g = (const unsigned int*)proj;
    unsigned int* s = (unsigned int*)sp;
    for (int idx = tid; idx < 3 * 64 * 32; idx += 128) {
      int t   = idx >> 11;
      int rem = idx & 2047;
      int r   = rem >> 5;
      int e2  = rem & 31;
      s[(t * 64 + r) * (PITCH / 2) + e2] =
          g[(size_t)t * (BATCH * RED / 2) + (size_t)(row_base + r) * (RED / 2) + e2];
    }
  }

  // ---- async staging state: 1024 x 16B per slab / 128 thr = 8 items each ----
  unsigned long long ga[8];
  unsigned int       lbs[8];
#pragma unroll
  for (int i = 0; i < 8; ++i) {
    int idx = tid + 128 * i;
    int col = idx >> 3;          // 0..127
    int seg = idx & 7;           // 16B segment within 128B K-row
    ga[i]  = (unsigned long long)(uintptr_t)
             (w1bf + (size_t)(col_base + col) * IN_DIM + seg * 8);
    lbs[i] = (unsigned int)(uintptr_t)&sb[0][col][seg * 8];
  }

#define ISSUE_STAGE(bufsel)                                                 \
  {                                                                         \
    const unsigned int boff = (unsigned int)(bufsel) * (unsigned int)sizeof(sb[0]); \
    _Pragma("unroll")                                                       \
    for (int i = 0; i < 8; ++i) {                                           \
      asm volatile("global_load_async_to_lds_b128 %0, %1, off"              \
                   :: "v"(lbs[i] + boff), "v"(ga[i]) : "memory");           \
      ga[i] += 128;  /* advance 64 bf16 K elements */                       \
    }                                                                       \
  }

  ISSUE_STAGE(0);                 // prologue: slab for k0 = 0 into buffer 0

  const int wave  = tid >> 5;
  const int lane  = tid & 31;
  const int wm    = wave & 1;        // 2 waves in M (32 rows each)
  const int wn    = wave >> 1;       // 2 waves in N (64 cols each)
  const int lrow  = lane & 15;
  const int khalf = lane >> 4;       // 0: K 0-7/16-23, 1: K 8-15/24-31
  const int kbase = khalf * 8;

  v8f acc[2][4] = {};

  for (int it = 0; it < NIT; ++it) {
    const int k0  = it * 64;
    const int cur = it & 1;

    asm volatile("s_wait_asynccnt 0x0" ::: "memory");  // my loads into cur done
    __syncthreads();                                   // everyone's done; nxt free

    if (it + 1 < NIT) ISSUE_STAGE(1 - cur);            // prefetch next slab

    const int blk = k0 >> 12;        // 0: i*j, 1: i*k, 2: j*k
    const int d   = (k0 & 4095) >> 6;

    const unsigned short (*L)[PITCH] = sp[blk == 2 ? 1 : 0];
    const unsigned short (*R)[PITCH] = sp[blk == 0 ? 1 : 2];

    // left scalar per M-tile, splat into both bf16 halves (reused 2 halves)
    unsigned int lv2[2];
#pragma unroll
    for (int mt = 0; mt < 2; ++mt) {
      unsigned int u = L[wm * 32 + mt * 16 + lrow][d];
      lv2[mt] = u | (u << 16);
    }

#pragma unroll
    for (int half = 0; half < 2; ++half) {
      const int e0 = half * 32;

      // ---- A fragments: 2 x b128 LDS load + 8 pk_mul per tile ----
      BFrag afrag[2];
#pragma unroll
      for (int mt = 0; mt < 2; ++mt) {
        const int r = wm * 32 + mt * 16 + lrow;
        const uint4 ra = *(const uint4*)&R[r][e0 + kbase];        // K +0..7
        const uint4 rb = *(const uint4*)&R[r][e0 + kbase + 16];   // K +16..23
        afrag[mt].u[0] = pk_mul_bf16(lv2[mt], ra.x);
        afrag[mt].u[1] = pk_mul_bf16(lv2[mt], ra.y);
        afrag[mt].u[2] = pk_mul_bf16(lv2[mt], ra.z);
        afrag[mt].u[3] = pk_mul_bf16(lv2[mt], ra.w);
        afrag[mt].u[4] = pk_mul_bf16(lv2[mt], rb.x);
        afrag[mt].u[5] = pk_mul_bf16(lv2[mt], rb.y);
        afrag[mt].u[6] = pk_mul_bf16(lv2[mt], rb.z);
        afrag[mt].u[7] = pk_mul_bf16(lv2[mt], rb.w);
      }

      // ---- B fragments from the staged slab: 2 x ds_load_b128 per tile ----
      BFrag bfrag[4];
#pragma unroll
      for (int nt = 0; nt < 4; ++nt) {
        const int cl = wn * 64 + nt * 16 + lrow;
        const unsigned short* wp = &sb[cur][cl][e0 + kbase];
        bfrag[nt].q[0] = *(const uint4*)wp;          // K kbase..kbase+7
        bfrag[nt].q[1] = *(const uint4*)(wp + 16);   // K kbase+16..kbase+23
      }

      // ---- 8 WMMAs per wave per 32-wide K-step ----
#pragma unroll
      for (int mt = 0; mt < 2; ++mt)
#pragma unroll
        for (int nt = 0; nt < 4; ++nt)
          acc[mt][nt] = __builtin_amdgcn_wmma_f32_16x16x32_bf16(
              false, afrag[mt].v, false, bfrag[nt].v,
              (short)0, acc[mt][nt], false, false);
    }
  }
#undef ISSUE_STAGE

  // ---- epilogue: + b1, ReLU, store (C layout: N=lane&15, M=r + 8*khalf) ----
#pragma unroll
  for (int mt = 0; mt < 2; ++mt) {
#pragma unroll
    for (int nt = 0; nt < 4; ++nt) {
      const int col  = col_base + wn * 64 + nt * 16 + lrow;
      const float bb = b1[col];
      const int m0   = row_base + wm * 32 + mt * 16 + khalf * 8;
#pragma unroll
      for (int r = 0; r < 8; ++r) {
        float v = acc[mt][nt][r] + bb;
        v = v > 0.0f ? v : 0.0f;
        h1[(size_t)(m0 + r) * H1_DIM + col] = v;
      }
    }
  }
}

// ---------------------------------------------------------------------------
// Kernel 4: h2 = tanh(h1 @ W2^T + b2)   (0.5 GFLOP, fp32 VALU)
// ---------------------------------------------------------------------------
__global__ __launch_bounds__(256) void k_layer2(
    const float* __restrict__ h1, const float* __restrict__ W2,
    const float* __restrict__ b2, float* __restrict__ h2) {
  const int tid = threadIdx.x;
  const int b = blockIdx.x * 2 + (tid >> 7);
  const int n = tid & 127;
  const float4* x = (const float4*)(h1 + (size_t)b * H1_DIM);
  const float4* w = (const float4*)(W2 + (size_t)n * H1_DIM);
  float acc = b2[n];
#pragma unroll 4
  for (int k = 0; k < H1_DIM / 4; ++k) {
    float4 xv = x[k], wv = w[k];
    acc = fmaf(xv.x, wv.x, acc);
    acc = fmaf(xv.y, wv.y, acc);
    acc = fmaf(xv.z, wv.z, acc);
    acc = fmaf(xv.w, wv.w, acc);
  }
  h2[(size_t)b * H2_DIM + n] = tanhf(acc);
}

// ---------------------------------------------------------------------------
// Kernel 5: out = h2 @ W3^T + b3 — one wave32 per row, shfl_xor reduction.
// ---------------------------------------------------------------------------
__global__ __launch_bounds__(256) void k_layer3(
    const float* __restrict__ h2, const float* __restrict__ W3,
    const float* __restrict__ b3, float* __restrict__ out) {
  const int wave = threadIdx.x >> 5;
  const int lane = threadIdx.x & 31;
  const int b = blockIdx.x * 8 + wave;
  const float4 xv = *(const float4*)(h2 + (size_t)b * H2_DIM + lane * 4);
  const float4 wv = *(const float4*)(W3 + lane * 4);
  float s = xv.x * wv.x + xv.y * wv.y + xv.z * wv.z + xv.w * wv.w;
#pragma unroll
  for (int off = 16; off > 0; off >>= 1)
    s += __shfl_xor(s, off, 32);
  if (lane == 0) out[b] = s + b3[0];
}

// ---------------------------------------------------------------------------
extern "C" void kernel_launch(void* const* d_in, const int* in_sizes, int n_in,
                              void* d_out, int out_size, void* d_ws, size_t ws_size,
                              hipStream_t stream) {
  const int*   i_idx = (const int*)d_in[0];
  const int*   j_idx = (const int*)d_in[1];
  const int*   k_idx = (const int*)d_in[2];
  const float* uemb  = (const float*)d_in[3];
  const float* iemb  = (const float*)d_in[4];
  const float* temb  = (const float*)d_in[5];
  const float* Wi = (const float*)d_in[6];
  const float* bi = (const float*)d_in[7];
  const float* Wj = (const float*)d_in[8];
  const float* bj = (const float*)d_in[9];
  const float* Wk = (const float*)d_in[10];
  const float* bk = (const float*)d_in[11];
  const float* W1 = (const float*)d_in[12];
  const float* b1 = (const float*)d_in[13];
  const float* W2 = (const float*)d_in[14];
  const float* b2 = (const float*)d_in[15];
  const float* W3 = (const float*)d_in[16];
  const float* b3 = (const float*)d_in[17];
  float* out = (float*)d_out;

  // workspace layout (~21 MB): bf16 projections | h1 | h2 | bf16 W1
  char* wsb = (char*)d_ws;
  unsigned short* proj = (unsigned short*)wsb;                 // [3][8192][64] bf16
  float* h1 = (float*)(wsb + (size_t)3 * BATCH * RED * 2);     // [8192][256] f32
  float* h2 = (float*)((char*)h1 + (size_t)BATCH * H1_DIM * 4);// [8192][128] f32
  unsigned short* w1bf =
      (unsigned short*)((char*)h2 + (size_t)BATCH * H2_DIM * 4);

  k_w1_to_bf16<<<(H1_DIM * IN_DIM + 255) / 256, 256, 0, stream>>>(
      W1, w1bf, H1_DIM * IN_DIM);

  k_gather_proj<<<BATCH, 64, 0, stream>>>(
      i_idx, j_idx, k_idx, uemb, iemb, temb,
      Wi, bi, Wj, bj, Wk, bk,
      proj, proj + (size_t)BATCH * RED, proj + (size_t)2 * BATCH * RED);

  k_layer1_wmma<<<(BATCH / 64) * 2, 128, 0, stream>>>(proj, w1bf, b1, h1);

  k_layer2<<<BATCH / 2, 256, 0, stream>>>(h1, W2, b2, h2);

  k_layer3<<<BATCH / 8, 256, 0, stream>>>(h2, W3, b3, out);
}
